// StateEliminationNNet_16432544874681
// MI455X (gfx1250) — matrix operand order repaired and checked by reading, not compiled, not run on hardware
//
#include <hip/hip_runtime.h>
#include <hip/hip_bf16.h>

typedef __attribute__((ext_vector_type(16))) _Float16 v16h;
typedef __attribute__((ext_vector_type(8)))  float    v8f;

union V16H { v16h v; _Float16 h[16]; int4 q[2]; };

__device__ inline v8f wmma16(v16h a, v16h b, v8f c) {
    return __builtin_amdgcn_wmma_f32_16x16x32_f16(
        /*neg_a=*/false, a, /*neg_b=*/false, b,
        /*c_mod=*/(short)0, c, /*reuse_a=*/false, /*reuse_b=*/false);
}

__device__ inline v16h load_v16h_2seg(const _Float16* p0, const _Float16* p1) {
    V16H u;
    u.q[0] = *(const int4*)p0;
    u.q[1] = *(const int4*)p1;
    return u.v;
}

// Fast transcendentals: v_exp_f32 + v_rcp_f32, branch-free.
__device__ inline float fast_sigmoid(float x) {
    return __builtin_amdgcn_rcpf(1.0f + __expf(-x));
}
__device__ inline float fast_tanh(float x) {
    return 1.0f - 2.0f * __builtin_amdgcn_rcpf(1.0f + __expf(2.0f * x));
}

// ---------------------------------------------------------------------------
// proj[v][g] = b[g] + sum_e emb[v][e] * w_ih[g][e]   (20 x 128)
// ---------------------------------------------------------------------------
__global__ void proj_kernel(const float* __restrict__ emb,
                            const float* __restrict__ wih,
                            const float* __restrict__ bias,
                            float* __restrict__ proj) {
    int i = blockIdx.x * blockDim.x + threadIdx.x;
    if (i >= 20 * 128) return;
    int v = i >> 7, g = i & 127;
    float a = bias[g];
#pragma unroll
    for (int e = 0; e < 8; ++e) a += emb[v * 8 + e] * wih[g * 8 + e];
    proj[i] = a;
}

// ---------------------------------------------------------------------------
// Pack f32 weight matrix into WMMA 16-bit B fragments (32xK x 16xN tiles).
// B element (k, n) = W[k*sk + n*sn]; k >= K padded with 0.
// ---------------------------------------------------------------------------
__global__ void pack_b_kernel(const float* __restrict__ W, _Float16* __restrict__ out,
                              int K, int sk, int sn, int kchunks, int ntiles) {
    int t = blockIdx.x * blockDim.x + threadIdx.x;
    int total = kchunks * ntiles * 32;
    if (t >= total) return;
    int lane = t & 31, frag = t >> 5;
    int nt = frag % ntiles, kc = frag / ntiles;
    int n = nt * 16 + (lane & 15);
    V16H u;
#pragma unroll
    for (int v = 0; v < 8; ++v) {
        int k = kc * 32 + ((v < 4) ? 2 * v : 16 + 2 * (v - 4)) + ((lane & 16) ? 8 : 0);
        u.h[2 * v]     = (_Float16)((k     < K) ? W[(size_t)k       * sk + (size_t)n * sn] : 0.0f);
        u.h[2 * v + 1] = (_Float16)((k + 1 < K) ? W[(size_t)(k + 1) * sk + (size_t)n * sn] : 0.0f);
    }
    *(v16h*)(out + (size_t)frag * 512 + lane * 16) = u.v;
}

// ---------------------------------------------------------------------------
// BiLSTM direction pass. One wave handles 16 edges; per step:
//   z = proj[token] (gathered into C layout) + h @ W_hh^T  (8 WMMA tiles)
//   gate nonlinearity, c/h update, h -> LDS -> A-layout reload.
// ---------------------------------------------------------------------------
__global__ __launch_bounds__(256) void lstm_kernel(
    const int* __restrict__ tokens,        // [E][50]
    const float* __restrict__ proj,        // [20][128]
    const _Float16* __restrict__ whh_pack, // 8 B-fragments
    float* __restrict__ regex_feat,        // [E][64]
    int E, int reverse, int col_off) {
    __shared__ float s_proj[20 * 128];
    __shared__ __align__(16) _Float16 s_h[8][16 * 32];
    __shared__ int s_tok[8][50 * 16];

    int tid = threadIdx.x, w = tid >> 5, lane = tid & 31;
    for (int i = tid; i < 20 * 128; i += 256) s_proj[i] = proj[i];
    __syncthreads();

    int waveg = blockIdx.x * 8 + w;
    if (waveg * 16 >= E) return;
    size_t ebase = (size_t)waveg * 16;

    for (int i = lane; i < 800; i += 32) {
        int m = i / 50, t = i % 50;
        s_tok[w][t * 16 + m] = tokens[(ebase + m) * 50 + t];
    }
    v16h Bm[8];
#pragma unroll
    for (int n = 0; n < 8; ++n)
        Bm[n] = *(const v16h*)(whh_pack + (size_t)n * 512 + lane * 16);

    int nl = lane & 15;
    int mb = (lane & 16) ? 8 : 0;
    int kb = (lane & 16) ? 8 : 0;

    v16h Ah;
#pragma unroll
    for (int i = 0; i < 16; ++i) Ah[i] = (_Float16)0.0f;
    v8f cst[2], acc[2];
#pragma unroll
    for (int j = 0; j < 2; ++j)
#pragma unroll
        for (int r = 0; r < 8; ++r) { cst[j][r] = 0.0f; acc[j][r] = 0.0f; }

    asm volatile("s_wait_dscnt 0x0" ::: "memory");   // token stores visible

    for (int t = 0; t < 50; ++t) {
        int tt = reverse ? (49 - t) : t;
        const int* trow = &s_tok[w][tt * 16];
        int tk[8];
#pragma unroll
        for (int r = 0; r < 8; ++r) tk[r] = trow[mb + r];

#pragma unroll
        for (int j = 0; j < 2; ++j) {               // hidden halves 0-15 / 16-31
            v8f z[4];
#pragma unroll
            for (int g = 0; g < 4; ++g) {           // i, f, g, o
                int n = g * 2 + j;
                v8f c;
#pragma unroll
                for (int r = 0; r < 8; ++r)
                    c[r] = s_proj[tk[r] * 128 + n * 16 + nl];
                z[g] = wmma16(Ah, Bm[n], c);
            }
#pragma unroll
            for (int r = 0; r < 8; ++r) {
                float cv = fast_sigmoid(z[1][r]) * cst[j][r] +
                           fast_sigmoid(z[0][r]) * fast_tanh(z[2][r]);
                cst[j][r] = cv;
                float hv = fast_sigmoid(z[3][r]) * fast_tanh(cv);
                acc[j][r] += hv;
                s_h[w][(mb + r) * 32 + j * 16 + nl] = (_Float16)hv;
            }
        }
        asm volatile("s_wait_dscnt 0x0" ::: "memory"); // h stores before reload
        const _Float16* hr = &s_h[w][nl * 32 + kb];
        Ah = load_v16h_2seg(hr, hr + 16);
    }
#pragma unroll
    for (int j = 0; j < 2; ++j)
#pragma unroll
        for (int r = 0; r < 8; ++r)
            regex_feat[(ebase + mb + r) * 64 + col_off + j * 16 + nl] =
                acc[j][r] * 0.02f;   // mean over T=50
}

// ---------------------------------------------------------------------------
// Edge scatter: atomic mean-pool numerators + counts. One wave per edge.
// ---------------------------------------------------------------------------
__global__ __launch_bounds__(256) void scatter_kernel(
    const int* __restrict__ eidx, const float* __restrict__ src_num,
    const float* __restrict__ tgt_num, const float* __restrict__ regex_feat,
    float* __restrict__ out_acc, float* __restrict__ in_acc,
    float* __restrict__ cnt_out, float* __restrict__ cnt_in, int E) {
    int w = (blockIdx.x * blockDim.x + threadIdx.x) >> 5;
    int lane = threadIdx.x & 31;
    if (w >= E) return;
    int s = eidx[w], t = eidx[E + w];
#pragma unroll
    for (int c = 0; c < 4; ++c) {
        int f = lane + c * 32;
        if (f < 117) {
            float rf = (f >= 53) ? regex_feat[(size_t)w * 64 + (f - 53)] : 0.0f;
            float ov = (f < 53) ? tgt_num[(size_t)w * 53 + f] : rf;
            float iv = (f < 53) ? src_num[(size_t)w * 53 + f] : rf;
            atomicAdd(out_acc + (size_t)s * 117 + f, ov);
            atomicAdd(in_acc  + (size_t)t * 117 + f, iv);
        }
    }
    if (lane == 0) {
        atomicAdd(cnt_out + s, 1.0f);
        atomicAdd(cnt_in  + t, 1.0f);
    }
}

// ---------------------------------------------------------------------------
// xc16[node][0:320] = f16( [x(55) | in_tr(117) | out_tr(117) | zeros(31)] )
// ---------------------------------------------------------------------------
__global__ void xc_kernel(const float* __restrict__ x,
                          const float* __restrict__ in_acc, const float* __restrict__ cnt_in,
                          const float* __restrict__ out_acc, const float* __restrict__ cnt_out,
                          _Float16* __restrict__ xc16, int N) {
    size_t node = blockIdx.x;
    int f = threadIdx.x;
    float v;
    if (f < 55)       v = x[node * 55 + f];
    else if (f < 172) v = in_acc[node * 117 + (f - 55)]  * __builtin_amdgcn_rcpf(fmaxf(cnt_in[node], 1.0f));
    else if (f < 289) v = out_acc[node * 117 + (f - 172)] * __builtin_amdgcn_rcpf(fmaxf(cnt_out[node], 1.0f));
    else              v = 0.0f;
    xc16[node * 320 + f] = (_Float16)v;
}

// ---------------------------------------------------------------------------
// Graph mean-pool s[g][0:289] (batch is contiguous, 32 nodes/graph).
// ---------------------------------------------------------------------------
__global__ void spool_kernel(const float* __restrict__ x,
                             const float* __restrict__ in_acc, const float* __restrict__ cnt_in,
                             const float* __restrict__ out_acc, const float* __restrict__ cnt_out,
                             float* __restrict__ s, int Bg) {
    int g = blockIdx.x;
    for (int f = threadIdx.x; f < 289; f += blockDim.x) {
        float a = 0.0f;
        for (int m = 0; m < 32; ++m) {
            size_t node = (size_t)g * 32 + m;
            float v;
            if (f < 55)       v = x[node * 55 + f];
            else if (f < 172) v = in_acc[node * 117 + (f - 55)]  / fmaxf(cnt_in[node], 1.0f);
            else              v = out_acc[node * 117 + (f - 172)] / fmaxf(cnt_out[node], 1.0f);
            a += v;
        }
        s[(size_t)g * 289 + f] = a * (1.0f / 32.0f);
    }
}

// ---------------------------------------------------------------------------
// v head: v[g] = relu(s[g] @ v1_w + v1_b) @ v2_w + v2_b. One wave per graph.
// ---------------------------------------------------------------------------
__global__ __launch_bounds__(256) void vhead_kernel(
    const float* __restrict__ s, const float* __restrict__ v1w,
    const float* __restrict__ v1b, const float* __restrict__ v2w,
    const float* __restrict__ v2b, float* __restrict__ vout, int Bg) {
    int g = (blockIdx.x * blockDim.x + threadIdx.x) >> 5;
    int lane = threadIdx.x & 31;
    if (g >= Bg) return;
    const float* sg = s + (size_t)g * 289;
    float a = v1b[lane];
    for (int k = 0; k < 289; ++k) a += sg[k] * v1w[k * 32 + lane];
    a = fmaxf(a, 0.0f) * v2w[lane];
#pragma unroll
    for (int off = 16; off; off >>= 1) a += __shfl_xor(a, off, 32);
    if (lane == 0) vout[g] = a + v2b[0];
}

// ---------------------------------------------------------------------------
// Fused pi head: 320 -> 128 -> 64 -> 32 -> 1 (all WMMA except last).
// ---------------------------------------------------------------------------
__global__ __launch_bounds__(256) void pi_kernel(
    const _Float16* __restrict__ xc16,
    const _Float16* __restrict__ wp1, const float* __restrict__ p1b,
    const _Float16* __restrict__ wp2, const float* __restrict__ p2b,
    const _Float16* __restrict__ wp3, const float* __restrict__ p3b,
    const float* __restrict__ p4w, const float* __restrict__ p4b,
    float* __restrict__ pi_out, int N) {
    __shared__ __align__(16) _Float16 s_h1[8][16 * 128];
    __shared__ __align__(16) _Float16 s_h2[8][16 * 64];
    __shared__ __align__(16) _Float16 s_h3[8][16 * 32];
    int tid = threadIdx.x, w = tid >> 5, lane = tid & 31;
    int nl = lane & 15;
    int mb = (lane & 16) ? 8 : 0;
    int kb = (lane & 16) ? 8 : 0;
    size_t nodebase = ((size_t)blockIdx.x * 8 + w) * 16;
    if (nodebase >= (size_t)N) return;

    // Layer 1: K=320 (289 padded), N=128
    v8f C1[8];
#pragma unroll
    for (int nt = 0; nt < 8; ++nt) {
        float b = p1b[nt * 16 + nl];
#pragma unroll
        for (int r = 0; r < 8; ++r) C1[nt][r] = b;
    }
    const _Float16* arow = xc16 + (nodebase + nl) * 320;
#pragma unroll
    for (int kc = 0; kc < 10; ++kc) {
        v16h A = load_v16h_2seg(arow + kc * 32 + kb, arow + kc * 32 + kb + 16);
#pragma unroll
        for (int nt = 0; nt < 8; ++nt) {
            v16h Bf = *(const v16h*)(wp1 + (size_t)(kc * 8 + nt) * 512 + lane * 16);
            C1[nt] = wmma16(A, Bf, C1[nt]);
        }
    }
#pragma unroll
    for (int nt = 0; nt < 8; ++nt)
#pragma unroll
        for (int r = 0; r < 8; ++r)
            s_h1[w][(mb + r) * 128 + nt * 16 + nl] = (_Float16)fmaxf(C1[nt][r], 0.0f);
    asm volatile("s_wait_dscnt 0x0" ::: "memory");

    // Layer 2: K=128, N=64
    v8f C2[4];
#pragma unroll
    for (int nt = 0; nt < 4; ++nt) {
        float b = p2b[nt * 16 + nl];
#pragma unroll
        for (int r = 0; r < 8; ++r) C2[nt][r] = b;
    }
#pragma unroll
    for (int kc = 0; kc < 4; ++kc) {
        const _Float16* hr = &s_h1[w][nl * 128 + kc * 32 + kb];
        v16h A = load_v16h_2seg(hr, hr + 16);
#pragma unroll
        for (int nt = 0; nt < 4; ++nt) {
            v16h Bf = *(const v16h*)(wp2 + (size_t)(kc * 4 + nt) * 512 + lane * 16);
            C2[nt] = wmma16(A, Bf, C2[nt]);
        }
    }
#pragma unroll
    for (int nt = 0; nt < 4; ++nt)
#pragma unroll
        for (int r = 0; r < 8; ++r)
            s_h2[w][(mb + r) * 64 + nt * 16 + nl] = (_Float16)fmaxf(C2[nt][r], 0.0f);
    asm volatile("s_wait_dscnt 0x0" ::: "memory");

    // Layer 3: K=64, N=32
    v8f C3[2];
#pragma unroll
    for (int nt = 0; nt < 2; ++nt) {
        float b = p3b[nt * 16 + nl];
#pragma unroll
        for (int r = 0; r < 8; ++r) C3[nt][r] = b;
    }
#pragma unroll
    for (int kc = 0; kc < 2; ++kc) {
        const _Float16* hr = &s_h2[w][nl * 64 + kc * 32 + kb];
        v16h A = load_v16h_2seg(hr, hr + 16);
#pragma unroll
        for (int nt = 0; nt < 2; ++nt) {
            v16h Bf = *(const v16h*)(wp3 + (size_t)(kc * 2 + nt) * 512 + lane * 16);
            C3[nt] = wmma16(A, Bf, C3[nt]);
        }
    }
#pragma unroll
    for (int nt = 0; nt < 2; ++nt)
#pragma unroll
        for (int r = 0; r < 8; ++r)
            s_h3[w][(mb + r) * 32 + nt * 16 + nl] = (_Float16)fmaxf(C3[nt][r], 0.0f);
    asm volatile("s_wait_dscnt 0x0" ::: "memory");

    // Layer 4: K=32, N=1 (VALU dot, lanes 0-15 -> nodes 0-15 of the tile)
    if (lane < 16) {
        float a = p4b[0];
#pragma unroll
        for (int c = 0; c < 32; ++c)
            a += (float)s_h3[w][lane * 32 + c] * p4w[c];
        pi_out[nodebase + lane] = a;
    }
}

// ---------------------------------------------------------------------------
// Ragged pack + log_softmax. Columns 0..31 carry pi, 32..52 are -999.
// ---------------------------------------------------------------------------
__global__ __launch_bounds__(256) void logits_kernel(const float* __restrict__ pi,
                                                     float* __restrict__ out, int Bg) {
    int g = (blockIdx.x * blockDim.x + threadIdx.x) >> 5;
    int lane = threadIdx.x & 31;
    if (g >= Bg) return;
    float v0 = pi[(size_t)g * 32 + lane];
    float v1 = (lane < 21) ? -999.0f : -3.0e38f;
    float m = fmaxf(v0, v1);
#pragma unroll
    for (int off = 16; off; off >>= 1) m = fmaxf(m, __shfl_xor(m, off, 32));
    float e = __expf(v0 - m) + ((lane < 21) ? __expf(-999.0f - m) : 0.0f);
#pragma unroll
    for (int off = 16; off; off >>= 1) e += __shfl_xor(e, off, 32);
    float lse = m + __logf(e);
    out[(size_t)g * 53 + lane] = v0 - lse;
    if (lane < 21) out[(size_t)g * 53 + 32 + lane] = -999.0f - lse;
}

// ---------------------------------------------------------------------------
extern "C" void kernel_launch(void* const* d_in, const int* in_sizes, int n_in,
                              void* d_out, int out_size, void* d_ws, size_t ws_size,
                              hipStream_t stream) {
    const float* x       = (const float*)d_in[0];
    const float* src_num = (const float*)d_in[1];
    const float* tgt_num = (const float*)d_in[2];
    const int*   tokens  = (const int*)d_in[3];
    const int*   eidx    = (const int*)d_in[4];
    const float* emb     = (const float*)d_in[6];
    const float* w_ih_f  = (const float*)d_in[7];
    const float* w_hh_f  = (const float*)d_in[8];
    const float* b_f     = (const float*)d_in[9];
    const float* w_ih_b  = (const float*)d_in[10];
    const float* w_hh_b  = (const float*)d_in[11];
    const float* b_b     = (const float*)d_in[12];
    const float* p1_w = (const float*)d_in[13]; const float* p1_b = (const float*)d_in[14];
    const float* p2_w = (const float*)d_in[15]; const float* p2_b = (const float*)d_in[16];
    const float* p3_w = (const float*)d_in[17]; const float* p3_b = (const float*)d_in[18];
    const float* p4_w = (const float*)d_in[19]; const float* p4_b = (const float*)d_in[20];
    const float* v1_w = (const float*)d_in[21]; const float* v1_b = (const float*)d_in[22];
    const float* v2_w = (const float*)d_in[23]; const float* v2_b = (const float*)d_in[24];
    (void)n_in; (void)out_size; (void)ws_size;

    const int N  = in_sizes[0] / 55;   // 131072 nodes
    const int E  = in_sizes[1] / 53;   // 65536 edges
    const int Bg = N / 32;             // 4096 graphs

    char* ws = (char*)d_ws;
    size_t off = 0;
    auto take = [&](size_t bytes) {
        size_t r = off;
        off = (off + bytes + 255) & ~(size_t)255;
        return r;
    };
    float*    proj_f     = (float*)(ws + take(2560 * 4));
    float*    proj_b     = (float*)(ws + take(2560 * 4));
    float*    regex_feat = (float*)(ws + take((size_t)E * 64 * 4));
    size_t acc_begin = off;
    float*    out_acc    = (float*)(ws + take((size_t)N * 117 * 4));
    float*    in_acc     = (float*)(ws + take((size_t)N * 117 * 4));
    float*    cnt_out    = (float*)(ws + take((size_t)N * 4));
    float*    cnt_in     = (float*)(ws + take((size_t)N * 4));
    size_t acc_end = off;
    _Float16* xc16       = (_Float16*)(ws + take((size_t)N * 320 * 2));
    float*    s_buf      = (float*)(ws + take((size_t)Bg * 289 * 4));
    float*    pi_buf     = (float*)(ws + take((size_t)N * 4));
    _Float16* wp_hf      = (_Float16*)(ws + take(8 * 1024));
    _Float16* wp_hb      = (_Float16*)(ws + take(8 * 1024));
    _Float16* wp1        = (_Float16*)(ws + take(80 * 1024));
    _Float16* wp2        = (_Float16*)(ws + take(16 * 1024));
    _Float16* wp3        = (_Float16*)(ws + take(4 * 1024));

    float* logits = (float*)d_out;
    float* vout   = (float*)d_out + (size_t)Bg * 53;

    hipMemsetAsync(ws + acc_begin, 0, acc_end - acc_begin, stream);

    proj_kernel<<<10, 256, 0, stream>>>(emb, w_ih_f, b_f, proj_f);
    proj_kernel<<<10, 256, 0, stream>>>(emb, w_ih_b, b_b, proj_b);
    pack_b_kernel<<<1, 256, 0, stream>>>(w_hh_f, wp_hf, 32, 1, 32, 1, 8);
    pack_b_kernel<<<1, 256, 0, stream>>>(w_hh_b, wp_hb, 32, 1, 32, 1, 8);
    pack_b_kernel<<<10, 256, 0, stream>>>(p1_w, wp1, 289, 128, 1, 10, 8);
    pack_b_kernel<<<2, 256, 0, stream>>>(p2_w, wp2, 128, 64, 1, 4, 4);
    pack_b_kernel<<<1, 256, 0, stream>>>(p3_w, wp3, 64, 32, 1, 2, 2);

    lstm_kernel<<<E / 128, 256, 0, stream>>>(tokens, proj_f, wp_hf, regex_feat, E, 0, 0);
    lstm_kernel<<<E / 128, 256, 0, stream>>>(tokens, proj_b, wp_hb, regex_feat, E, 1, 32);

    scatter_kernel<<<E / 8, 256, 0, stream>>>(eidx, src_num, tgt_num, regex_feat,
                                              out_acc, in_acc, cnt_out, cnt_in, E);

    xc_kernel<<<N, 320, 0, stream>>>(x, in_acc, cnt_in, out_acc, cnt_out, xc16, N);
    spool_kernel<<<Bg, 128, 0, stream>>>(x, in_acc, cnt_in, out_acc, cnt_out, s_buf, Bg);
    vhead_kernel<<<Bg / 8, 256, 0, stream>>>(s_buf, v1_w, v1_b, v2_w, v2_b, vout, Bg);

    pi_kernel<<<N / 128, 256, 0, stream>>>(xc16, wp1, p1_b, wp2, p2_b, wp3, p3_b,
                                           p4_w, p4_b, pi_buf, N);
    logits_kernel<<<Bg / 8, 256, 0, stream>>>(pi_buf, logits, Bg);
}
